// LSTM_cell_60146722013778
// MI455X (gfx1250) — matrix operand order repaired and checked by reading, not compiled, not run on hardware
//
#include <hip/hip_runtime.h>
#include <hip/hip_bf16.h>
#include <math.h>

#define HIDDEN 1024
#define INPUT  1024
#define BS     32
#define SEQ    512
#define NGATE  4
#define NWG    64                 // persistent workgroups in recurrent kernel
#define NSLICE (HIDDEN / NWG)     // 16 hidden columns per workgroup
#define HPAD   1032               // padded row stride (bf16 elems): 516 dwords -> bank spread

typedef __attribute__((ext_vector_type(16))) __bf16 bf16x16;
typedef __attribute__((ext_vector_type(8)))  __bf16 bf16x8;
typedef __attribute__((ext_vector_type(8)))  float  f32x8;

// ---- WMMA fragment helpers (CDNA5 16-bit layouts, wave32) -------------------
// A (16x32 bf16): lane L holds row m=L&15; lanes 0-15: K={0..7,16..23},
// lanes 16-31: K={8..15,24..31}  -> two contiguous 16B chunks per lane.
__device__ __forceinline__ bf16x16 load_frag_a(const __bf16* rowptr, int k0, int hi) {
  union { bf16x16 v; bf16x8 h[2]; } u;
  const __bf16* p = rowptr + k0 + hi * 8;
  u.h[0] = *(const bf16x8*)(p);
  u.h[1] = *(const bf16x8*)(p + 16);
  return u.v;
}
// B (32x16 bf16): lane L holds column n=L&15; lanes 0-15: K={0..7,16..23},
// lanes 16-31: K={8..15,24..31}. W stored [n][k] -> two contiguous 16B chunks.
__device__ __forceinline__ bf16x16 load_frag_b(const __bf16* colptr, int k0, int hi) {
  union { bf16x16 v; bf16x8 h[2]; } u;
  const __bf16* p = colptr + k0 + hi * 8;
  u.h[0] = *(const bf16x8*)(p);
  u.h[1] = *(const bf16x8*)(p + 16);
  return u.v;
}

__device__ __forceinline__ f32x8 wmma_bf16(bf16x16 a, bf16x16 b, f32x8 c) {
  return __builtin_amdgcn_wmma_f32_16x16x32_bf16(false, a, false, b, (short)0, c, false, false);
}

__device__ __forceinline__ float sigmoidf_(float x) { return 1.0f / (1.0f + __expf(-x)); }

// ---- setup kernels ----------------------------------------------------------
__global__ void k_init(unsigned* __restrict__ arrive, __bf16* __restrict__ h_bf) {
  int i = blockIdx.x * 256 + threadIdx.x;
  if (i < SEQ) arrive[i] = 0u;
  if (i < BS * HIDDEN) h_bf[i] = (__bf16)0.0f;
}

__global__ void k_convert(const float* __restrict__ src, __bf16* __restrict__ dst, int n) {
  int i = blockIdx.x * 256 + threadIdx.x;
  if (i < n) dst[i] = (__bf16)src[i];
}

// transpose-convert one 1024x1024 f32 [k][n] matrix into bf16 [n][k]
__global__ void k_transpose_cvt(const float* __restrict__ src, __bf16* __restrict__ dst) {
  int d = blockIdx.x * 256 + threadIdx.x;   // k index (0..1023)
  int n = blockIdx.y;                       // n index (0..1023)
  dst[(size_t)n * INPUT + d] = (__bf16)src[(size_t)d * HIDDEN + n];
}

// ---- phase 1: G[gate][t*BS+b][n] = x @ U + bias  (M=16384, N=4x1024, K=1024)
// Software-pipelined: fragments for k+32 are in flight while k's WMMAs issue.
__global__ void __launch_bounds__(256)
k_xproj(const __bf16* __restrict__ xb, const __bf16* __restrict__ Ut,
        const float* __restrict__ bi, const float* __restrict__ bfp,
        const float* __restrict__ bg, const float* __restrict__ bo,
        float* __restrict__ G) {
  const int lane = threadIdx.x & 31;
  const int wave = threadIdx.x >> 5;
  const int hi   = lane >> 4;
  const int col  = lane & 15;

  int job   = blockIdx.x * 8 + wave;  // 65536 wave-jobs total
  int gate  = job >> 14;
  int rem   = job & 16383;
  int mtile = rem >> 4;
  int strip = rem & 15;
  int n0 = strip * 64;
  int m0 = mtile * 16;

  // A row for this lane: M ordered as m = t*BS + b; x_bf stored [b][t][d]
  int m = m0 + col;
  const __bf16* arow = xb + ((size_t)(m & (BS - 1)) * SEQ + (m >> 5)) * (size_t)INPUT;

  const __bf16* Ug = Ut + (size_t)gate * HIDDEN * INPUT;
  const __bf16* bp0 = Ug + (size_t)(n0 +  0 + col) * INPUT;
  const __bf16* bp1 = Ug + (size_t)(n0 + 16 + col) * INPUT;
  const __bf16* bp2 = Ug + (size_t)(n0 + 32 + col) * INPUT;
  const __bf16* bp3 = Ug + (size_t)(n0 + 48 + col) * INPUT;

  f32x8 acc0 = {}, acc1 = {}, acc2 = {}, acc3 = {};
  bf16x16 a  = load_frag_a(arow, 0, hi);
  bf16x16 f0 = load_frag_b(bp0, 0, hi);
  bf16x16 f1 = load_frag_b(bp1, 0, hi);
  bf16x16 f2 = load_frag_b(bp2, 0, hi);
  bf16x16 f3 = load_frag_b(bp3, 0, hi);
  #pragma unroll 4
  for (int k0 = 0; k0 < INPUT - 32; k0 += 32) {
    bf16x16 an  = load_frag_a(arow, k0 + 32, hi);
    bf16x16 fn0 = load_frag_b(bp0, k0 + 32, hi);
    bf16x16 fn1 = load_frag_b(bp1, k0 + 32, hi);
    bf16x16 fn2 = load_frag_b(bp2, k0 + 32, hi);
    bf16x16 fn3 = load_frag_b(bp3, k0 + 32, hi);
    acc0 = wmma_bf16(a, f0, acc0);
    acc1 = wmma_bf16(a, f1, acc1);
    acc2 = wmma_bf16(a, f2, acc2);
    acc3 = wmma_bf16(a, f3, acc3);
    a = an; f0 = fn0; f1 = fn1; f2 = fn2; f3 = fn3;
  }
  acc0 = wmma_bf16(a, f0, acc0);
  acc1 = wmma_bf16(a, f1, acc1);
  acc2 = wmma_bf16(a, f2, acc2);
  acc3 = wmma_bf16(a, f3, acc3);

  const float* bp = (gate == 0) ? bi : (gate == 1) ? bfp : (gate == 2) ? bg : bo;
  float* Gg = G + (size_t)gate * SEQ * BS * HIDDEN;
  #pragma unroll
  for (int nt = 0; nt < 4; ++nt) {
    int n = n0 + nt * 16 + col;
    float bias = bp[n];
    f32x8 acc = (nt == 0) ? acc0 : (nt == 1) ? acc1 : (nt == 2) ? acc2 : acc3;
    #pragma unroll
    for (int r = 0; r < 8; ++r) {
      int mm = m0 + r + hi * 8;               // C/D layout: M = r + hi*8
      Gg[(size_t)mm * HIDDEN + n] = acc[r] + bias;
    }
  }
}

// ---- phase 2: persistent recurrent kernel -----------------------------------
// 64 WGs x 128 threads (4 waves). Wave g computes gate g for this WG's
// 16-column hidden slice. W slice is LDS-resident for all 512 steps
// (~202 KB LDS of the 320 KB/WGP); cell state lives in registers.
__global__ void __launch_bounds__(128)
k_recurrent(const __bf16* __restrict__ Wt, const float* __restrict__ G,
            __bf16* __restrict__ h_bf, float* __restrict__ out_hs,
            float* __restrict__ out_h, float* __restrict__ out_c,
            unsigned* __restrict__ arrive) {
  __shared__ __align__(16) __bf16 h_lds[BS * HPAD];             // 66,048 B
  __shared__ __align__(16) __bf16 w_lds[NGATE * NSLICE * HPAD]; // 132,096 B
  __shared__ float xch[NGATE][2][8][32];                        // 8,192 B

  const int lane = threadIdx.x & 31;
  const int gate = threadIdx.x >> 5;  // wave id == gate id (i,f,g,o)
  const int hi   = lane >> 4;
  const int col  = lane & 15;
  const int n0   = blockIdx.x * NSLICE;
  const size_t GS = (size_t)SEQ * BS * HIDDEN;

  // ---- load this WG's W slice into LDS once (reused for all 512 steps) ----
  for (int i = threadIdx.x; i < NGATE * NSLICE * (HIDDEN / 8); i += 128) {
    int gc = i >> 7;            // gate*16 + col  (0..63)
    int ck = i & 127;           // 8-elem chunk within row
    int g2 = gc >> 4, c2 = gc & 15;
    *(bf16x8*)(w_lds + (size_t)gc * HPAD + ck * 8) =
        *(const bf16x8*)(Wt + ((size_t)g2 * HIDDEN + n0 + c2) * HIDDEN + ck * 8);
  }

  const __bf16* wcol = w_lds + (size_t)(gate * NSLICE + col) * HPAD;
  float c_s[8];
  #pragma unroll
  for (int r = 0; r < 8; ++r) c_s[r] = 0.0f;

  for (int t = 0; t < SEQ; ++t) {
    // stage full h (32x1024 bf16) into padded LDS rows
    for (int i = threadIdx.x; i < BS * (HIDDEN / 8); i += 128) {
      int row = i >> 7;
      int ck  = i & 127;
      *(bf16x8*)(h_lds + (size_t)row * HPAD + ck * 8) =
          *(const bf16x8*)(h_bf + (size_t)row * HIDDEN + ck * 8);
    }

    // waves 2,3 (idle during the later elementwise phase) prefetch this step's
    // G rows (HBM-resident, 256MB > L2) -> global_prefetch_b8, hiding HBM
    // latency behind the WMMA phase on the serial scan path.
    if (gate >= 2) {
      int g0 = (gate - 2) * 2;
      const float* pg0 = G + (size_t)g0 * GS + ((size_t)t * BS + lane) * HIDDEN + n0;
      const float* pg1 = G + (size_t)(g0 + 1) * GS + ((size_t)t * BS + lane) * HIDDEN + n0;
      __builtin_prefetch(pg0, 0, 3);
      __builtin_prefetch(pg1, 0, 3);
    }
    __syncthreads();

    const __bf16* a0row = h_lds + (size_t)col * HPAD;
    const __bf16* a1row = h_lds + (size_t)(16 + col) * HPAD;
    f32x8 acc0 = {}, acc1 = {};
    bf16x16 a0 = load_frag_a(a0row, 0, hi);
    bf16x16 a1 = load_frag_a(a1row, 0, hi);
    bf16x16 bv = load_frag_b(wcol, 0, hi);
    #pragma unroll 4
    for (int k0 = 0; k0 < HIDDEN - 32; k0 += 32) {
      bf16x16 a0n = load_frag_a(a0row, k0 + 32, hi);
      bf16x16 a1n = load_frag_a(a1row, k0 + 32, hi);
      bf16x16 bvn = load_frag_b(wcol, k0 + 32, hi);
      acc0 = wmma_bf16(a0, bv, acc0);
      acc1 = wmma_bf16(a1, bv, acc1);
      a0 = a0n; a1 = a1n; bv = bvn;
    }
    acc0 = wmma_bf16(a0, bv, acc0);
    acc1 = wmma_bf16(a1, bv, acc1);

    #pragma unroll
    for (int r = 0; r < 8; ++r) {
      xch[gate][0][r][lane] = acc0[r];
      xch[gate][1][r][lane] = acc1[r];
    }
    __syncthreads();

    // waves 0 and 1 each handle one m-tile of the element-wise cell update
    if (gate < 2) {
      const int mt = gate;
      int n = n0 + col;
      #pragma unroll
      for (int r = 0; r < 8; ++r) {
        int bb = mt * 16 + r + hi * 8;                     // batch row
        size_t gidx = ((size_t)t * BS + bb) * HIDDEN + n;
        float vi = xch[0][mt][r][lane] + G[0 * GS + gidx];
        float vf = xch[1][mt][r][lane] + G[1 * GS + gidx];
        float vg = xch[2][mt][r][lane] + G[2 * GS + gidx];
        float vo = xch[3][mt][r][lane] + G[3 * GS + gidx];
        float it = sigmoidf_(vi);
        float ft = sigmoidf_(vf);
        float gt = tanhf(vg);
        float ot = sigmoidf_(vo);
        float c = ft * c_s[r] + it * gt;
        c_s[r] = c;
        float h = ot * tanhf(c);
        out_hs[((size_t)bb * SEQ + t) * HIDDEN + n] = h;   // [bs, seq, hd]
        h_bf[bb * HIDDEN + n] = (__bf16)h;
        if (t == SEQ - 1) {
          out_h[bb * HIDDEN + n] = h;
          out_c[bb * HIDDEN + n] = c;
        }
      }
    }

    // device-wide barrier on per-timestep arrival counter
    __threadfence();
    __syncthreads();
    if (threadIdx.x == 0) {
      __hip_atomic_fetch_add(&arrive[t], 1u, __ATOMIC_ACQ_REL, __HIP_MEMORY_SCOPE_AGENT);
      while (__hip_atomic_load(&arrive[t], __ATOMIC_ACQUIRE, __HIP_MEMORY_SCOPE_AGENT) <
             (unsigned)NWG)
        __builtin_amdgcn_s_sleep(1);
    }
    __syncthreads();
  }
}

// ---- host launcher ----------------------------------------------------------
static inline size_t alignup(size_t x) { return (x + 255) & ~(size_t)255; }

extern "C" void kernel_launch(void* const* d_in, const int* in_sizes, int n_in,
                              void* d_out, int out_size, void* d_ws, size_t ws_size,
                              hipStream_t stream) {
  const float* x = (const float*)d_in[0];
  const float* Us[4] = {(const float*)d_in[1], (const float*)d_in[4],
                        (const float*)d_in[7], (const float*)d_in[10]};
  const float* Ws[4] = {(const float*)d_in[2], (const float*)d_in[5],
                        (const float*)d_in[8], (const float*)d_in[11]};
  const float* Bs[4] = {(const float*)d_in[3], (const float*)d_in[6],
                        (const float*)d_in[9], (const float*)d_in[12]};

  // workspace carve (~319 MB)
  char* p = (char*)d_ws;
  __bf16* xb = (__bf16*)p;      p += alignup((size_t)BS * SEQ * INPUT * 2);
  __bf16* Ut = (__bf16*)p;      p += alignup((size_t)NGATE * HIDDEN * INPUT * 2);
  __bf16* Wt = (__bf16*)p;      p += alignup((size_t)NGATE * HIDDEN * HIDDEN * 2);
  __bf16* hbf = (__bf16*)p;     p += alignup((size_t)BS * HIDDEN * 2);
  unsigned* arrive = (unsigned*)p; p += alignup((size_t)SEQ * 4);
  float* G = (float*)p;         p += alignup((size_t)NGATE * SEQ * BS * HIDDEN * 4);

  float* out_hs = (float*)d_out;
  float* out_h  = out_hs + (size_t)BS * SEQ * HIDDEN;
  float* out_c  = out_h + (size_t)BS * HIDDEN;

  k_init<<<128, 256, 0, stream>>>(arrive, hbf);
  k_convert<<<(BS * SEQ * INPUT + 255) / 256, 256, 0, stream>>>(x, xb, BS * SEQ * INPUT);
  for (int g = 0; g < 4; ++g) {
    k_transpose_cvt<<<dim3(4, 1024), 256, 0, stream>>>(Us[g], Ut + (size_t)g * HIDDEN * INPUT);
    k_transpose_cvt<<<dim3(4, 1024), 256, 0, stream>>>(Ws[g], Wt + (size_t)g * HIDDEN * HIDDEN);
  }
  k_xproj<<<8192, 256, 0, stream>>>(xb, Ut, Bs[0], Bs[1], Bs[2], Bs[3], G);
  k_recurrent<<<NWG, 128, 0, stream>>>(Wt, G, hbf, out_hs, out_h, out_c, arrive);
}
